// MultiheadSelfAttention1D_55353538511251
// MI455X (gfx1250) — matrix operand order repaired and checked by reading, not compiled, not run on hardware
//
#include <hip/hip_runtime.h>

// ---------------------------------------------------------------------------
// MI455X (gfx1250) fused MHA: prep (transpose/convert to bf16) -> QKV+RoPE ->
// flash attention -> O projection. All GEMMs: v_wmma_f32_16x16x32_bf16.
// All fragments loaded as 128-bit chunks (layouts are 16B-contiguous).
// ---------------------------------------------------------------------------

typedef __attribute__((ext_vector_type(16))) __bf16 bf16x16;
typedef __attribute__((ext_vector_type(8)))  __bf16 bf16x8;
typedef __attribute__((ext_vector_type(4)))  __bf16 bf16x4;
typedef __attribute__((ext_vector_type(8)))  float  f32x8;

#define B_    2
#define C_    512
#define L_    4096
#define H_    8
#define KVH_  2
#define DH_   64
#define ROT_  32
#define NQ_   (H_ * DH_)          // 512
#define NKV_  (2 * KVH_ * DH_)    // 256
#define SCALE_ 0.125f             // 64^-0.5
// 10000^(-i/16) = exp(-i * ln(10000)/16)
#define NEG_LOG_THETA_O16 (-0.575646273248511f)

__device__ __forceinline__ f32x8 wmma_bf16(bf16x16 a, bf16x16 b, f32x8 c) {
  return __builtin_amdgcn_wmma_f32_16x16x32_bf16(false, a, false, b,
                                                 (short)0, c, false, false);
}

// Build a 16-element fragment from two contiguous 16-byte halves.
__device__ __forceinline__ bf16x16 ld_frag(const __bf16* p0, const __bf16* p1) {
  union { bf16x16 v; bf16x8 h[2]; } u;
  u.h[0] = *reinterpret_cast<const bf16x8*>(p0);
  u.h[1] = *reinterpret_cast<const bf16x8*>(p1);
  return u.v;
}

__device__ __forceinline__ float redmax16(float x) {
  x = fmaxf(x, __shfl_xor(x, 1, 32));
  x = fmaxf(x, __shfl_xor(x, 2, 32));
  x = fmaxf(x, __shfl_xor(x, 4, 32));
  x = fmaxf(x, __shfl_xor(x, 8, 32));
  return x;
}
__device__ __forceinline__ float redsum16(float x) {
  x += __shfl_xor(x, 1, 32);
  x += __shfl_xor(x, 2, 32);
  x += __shfl_xor(x, 4, 32);
  x += __shfl_xor(x, 8, 32);
  return x;
}

// ---------------------------------------------------------------------------
// Prep A: x (B,C,L) f32 -> Xt (B,L,C) bf16 via LDS-tiled transpose.
// ---------------------------------------------------------------------------
__global__ void transpose_x_kernel(const float* __restrict__ x,
                                   __bf16* __restrict__ Xt) {
  __shared__ float tile[32][33];
  const int b  = blockIdx.z;
  const int l0 = blockIdx.x * 32;
  const int c0 = blockIdx.y * 32;
  const int tx = threadIdx.x, ty = threadIdx.y;

  const float* xb = x + (size_t)b * C_ * L_;
#pragma unroll
  for (int i = ty; i < 32; i += 8)
    tile[i][tx] = xb[(size_t)(c0 + i) * L_ + l0 + tx];
  __syncthreads();
  __bf16* xo = Xt + (size_t)b * L_ * C_;
#pragma unroll
  for (int i = ty; i < 32; i += 8)
    xo[(size_t)(l0 + i) * C_ + c0 + tx] = (__bf16)tile[tx][i];
}

// Prep B: f32 -> bf16 elementwise (weights). n must be a multiple of 4.
__global__ void cvt_bf16_kernel(const float* __restrict__ src,
                                __bf16* __restrict__ dst, int n4) {
  const int i = blockIdx.x * 256 + threadIdx.x;
  if (i < n4) {
    const float4 f = *reinterpret_cast<const float4*>(src + 4 * i);
    bf16x4 o;
    o.x = (__bf16)f.x; o.y = (__bf16)f.y;
    o.z = (__bf16)f.z; o.w = (__bf16)f.w;
    *reinterpret_cast<bf16x4*>(dst + 4 * i) = o;
  }
}

// ---------------------------------------------------------------------------
// Kernel 1: QKV projection + RoPE. One wave owns a 16x64 output tile
// (4 WMMA accumulators, A fragment reused 4x per K-step). Each 64-wide tile
// is exactly one Q head, one K head, or one V head.
// Q,K stored bf16 [l][d]; V stored transposed [d][l].
// ---------------------------------------------------------------------------
__global__ void qkv_rope_kernel(const __bf16* __restrict__ Xt,
                                const __bf16* __restrict__ Wqb,
                                const __bf16* __restrict__ Wkvb,
                                __bf16* __restrict__ Qb,
                                __bf16* __restrict__ Kb,
                                __bf16* __restrict__ Vt) {
  const int lane  = threadIdx.x & 31;
  const int wave  = threadIdx.x >> 5;
  const int col   = lane & 15;
  const int lhalf = lane >> 4;

  const int m0 = blockIdx.x * 16;
  const int n0 = (blockIdx.y * 4 + wave) * 64;   // 0..704 step 64
  const int b  = blockIdx.z;

  const bool    isQ   = (n0 < NQ_);
  const __bf16* W     = isQ ? Wqb : Wkvb;
  const int     wrow0 = isQ ? n0 : (n0 - NQ_);

  const __bf16* xrow = Xt + (size_t)b * L_ * C_ + (size_t)(m0 + col) * C_;

  f32x8 acc[4] = {};
  for (int k0 = 0; k0 < C_; k0 += 32) {
    const bf16x16 a = ld_frag(xrow + k0 + (lhalf << 3),
                              xrow + k0 + 16 + (lhalf << 3));
#pragma unroll
    for (int t = 0; t < 4; ++t) {
      const __bf16* wrow =
          W + (size_t)(wrow0 + t * 16 + col) * C_ + k0 + (lhalf << 3);
      acc[t] = wmma_bf16(a, ld_frag(wrow, wrow + 16), acc[t]);
    }
  }

  // --- epilogue: RoPE on d<32 (pairs live in adjacent lanes) + store ---
  if (isQ) {
    const int h = n0 / DH_;
    __bf16* qout = Qb + (((size_t)b * H_ + h) * L_) * DH_;
#pragma unroll
    for (int t = 0; t < 4; ++t) {
#pragma unroll
      for (int v = 0; v < 8; ++v) {
        const int l = m0 + v + (lhalf << 3);
        const int d = t * 16 + col;
        float val = acc[t][v];
        const float pv = __shfl_xor(val, 1, 32);
        if (t < 2) {   // rotary half (compile-time per tile)
          const float inv_freq = __expf(NEG_LOG_THETA_O16 * (float)(d >> 1));
          float s, c;
          __sincosf((float)l * inv_freq, &s, &c);
          val = (d & 1) ? (val * c + pv * s) : (val * c - pv * s);
        }
        qout[(size_t)l * DH_ + d] = (__bf16)(val * SCALE_);
      }
    }
  } else {
    const int nn  = n0 - NQ_;        // 0,64,128,192
    const int kvh = nn / (2 * DH_);
    if ((nn & (2 * DH_ - 1)) == 0) {   // K head (rope, no scale)
      __bf16* kout = Kb + (((size_t)b * KVH_ + kvh) * L_) * DH_;
#pragma unroll
      for (int t = 0; t < 4; ++t) {
#pragma unroll
        for (int v = 0; v < 8; ++v) {
          const int l = m0 + v + (lhalf << 3);
          const int d = t * 16 + col;
          float val = acc[t][v];
          const float pv = __shfl_xor(val, 1, 32);
          if (t < 2) {
            const float inv_freq = __expf(NEG_LOG_THETA_O16 * (float)(d >> 1));
            float s, c;
            __sincosf((float)l * inv_freq, &s, &c);
            val = (d & 1) ? (val * c + pv * s) : (val * c - pv * s);
          }
          kout[(size_t)l * DH_ + d] = (__bf16)val;
        }
      }
    } else {                           // V head -> transposed [d][l]
      __bf16* vout = Vt + (((size_t)b * KVH_ + kvh) * DH_) * (size_t)L_;
#pragma unroll
      for (int t = 0; t < 4; ++t) {
#pragma unroll
        for (int v = 0; v < 8; ++v) {
          const int l = m0 + v + (lhalf << 3);
          const int d = t * 16 + col;
          vout[(size_t)d * L_ + l] = (__bf16)acc[t][v];
        }
      }
    }
  }
}

// ---------------------------------------------------------------------------
// Kernel 2: flash attention. 4 waves/block, each wave owns 16 query rows.
// Per 32-key step: 4 WMMAs for S, online softmax, P D->A re-layout through a
// private LDS slice, 4 WMMAs for P@V. All fragments are b128 loads.
// ---------------------------------------------------------------------------
__global__ void flash_attn_kernel(const __bf16* __restrict__ Qb,
                                  const __bf16* __restrict__ Kb,
                                  const __bf16* __restrict__ Vt,
                                  __bf16* __restrict__ Ob) {
  __shared__ __align__(16) __bf16 plds[4][16][32];

  const int lane  = threadIdx.x & 31;
  const int wave  = threadIdx.x >> 5;
  const int col   = lane & 15;
  const int lhalf = lane >> 4;

  const int h   = blockIdx.y;
  const int b   = blockIdx.z;
  const int kvh = h / (H_ / KVH_);
  const int m0  = blockIdx.x * 64 + wave * 16;

  const __bf16* qrow =
      Qb + (((size_t)b * H_ + h) * L_ + (m0 + col)) * DH_;
  const __bf16* kbase = Kb + (((size_t)b * KVH_ + kvh) * L_) * DH_;
  const __bf16* vbase = Vt + (((size_t)b * KVH_ + kvh) * DH_) * (size_t)L_;

  // Q fragments (d-blocks [0,32) and [32,64)) resident for the whole loop.
  bf16x16 qf[2];
#pragma unroll
  for (int kb = 0; kb < 2; ++kb) {
    const __bf16* p = qrow + kb * 32 + (lhalf << 3);
    qf[kb] = ld_frag(p, p + 16);
  }

  float m_run[8], l_run[8];
  f32x8 accO[4] = {};
#pragma unroll
  for (int v = 0; v < 8; ++v) { m_run[v] = -3.0e38f; l_run[v] = 0.0f; }

  for (int j0 = 0; j0 < L_; j0 += 32) {
    // ---- S = Q K^T for two 16-key tiles ----
    f32x8 s[2];
#pragma unroll
    for (int jt = 0; jt < 2; ++jt) {
      const __bf16* krow = kbase + (size_t)(j0 + jt * 16 + col) * DH_;
      const bf16x16 kf0 = ld_frag(krow + (lhalf << 3),
                                  krow + 16 + (lhalf << 3));
      const bf16x16 kf1 = ld_frag(krow + 32 + (lhalf << 3),
                                  krow + 48 + (lhalf << 3));
      f32x8 z = {};
      z = wmma_bf16(qf[0], kf0, z);
      z = wmma_bf16(qf[1], kf1, z);
      s[jt] = z;
    }

    // ---- online softmax (row r = v + 8*lane_half; 16 cols per half) ----
    float alpha[8];
#pragma unroll
    for (int v = 0; v < 8; ++v) {
      const float rm = redmax16(fmaxf(s[0][v], s[1][v]));
      const float mn = fmaxf(m_run[v], rm);
      const float a  = __expf(m_run[v] - mn);
      const float p0 = __expf(s[0][v] - mn);
      const float p1 = __expf(s[1][v] - mn);
      l_run[v] = l_run[v] * a + redsum16(p0 + p1);
      m_run[v] = mn;
      alpha[v] = a;
      s[0][v] = p0;
      s[1][v] = p1;
    }

    // ---- P: D-layout -> A-layout via this wave's LDS slice ----
#pragma unroll
    for (int v = 0; v < 8; ++v) {
      const int row = v + (lhalf << 3);
      plds[wave][row][col]      = (__bf16)s[0][v];
      plds[wave][row][16 + col] = (__bf16)s[1][v];
    }
    __builtin_amdgcn_wave_barrier();   // LDS in-order per wave; block reorder
    const __bf16* prow = &plds[wave][col][0];
    const bf16x16 pf = ld_frag(prow + (lhalf << 3),
                               prow + 16 + (lhalf << 3));
    __builtin_amdgcn_wave_barrier();

    // ---- O = diag(alpha)*O + P @ V over four 16-wide d tiles ----
#pragma unroll
    for (int t = 0; t < 4; ++t) {
      const __bf16* vrow =
          vbase + (size_t)(t * 16 + col) * L_ + j0 + (lhalf << 3);
      const bf16x16 vf = ld_frag(vrow, vrow + 16);
      f32x8 o = accO[t];
#pragma unroll
      for (int v = 0; v < 8; ++v) o[v] *= alpha[v];
      accO[t] = wmma_bf16(pf, vf, o);
    }
  }

  // ---- normalize and store O as bf16 [b][l][h*64+d] ----
  __bf16* obase = Ob + ((size_t)b * L_) * NQ_ + h * DH_;
#pragma unroll
  for (int t = 0; t < 4; ++t) {
#pragma unroll
    for (int v = 0; v < 8; ++v) {
      const int l = m0 + v + (lhalf << 3);
      const int d = t * 16 + col;
      obase[(size_t)l * NQ_ + d] = (__bf16)(accO[t][v] / l_run[v]);
    }
  }
}

// ---------------------------------------------------------------------------
// Kernel 3: output projection. out[b,c,l] = sum_n O[b,l,n] * Wo[c,n].
// One wave owns a 16x64 tile of the (L x C) result.
// ---------------------------------------------------------------------------
__global__ void out_proj_kernel(const __bf16* __restrict__ Ob,
                                const __bf16* __restrict__ Wob,
                                float* __restrict__ out) {
  const int lane  = threadIdx.x & 31;
  const int wave  = threadIdx.x >> 5;
  const int col   = lane & 15;
  const int lhalf = lane >> 4;

  const int m0 = blockIdx.x * 16;
  const int c0 = (blockIdx.y * 4 + wave) * 64;
  const int b  = blockIdx.z;

  const __bf16* orow =
      Ob + (size_t)b * L_ * NQ_ + (size_t)(m0 + col) * NQ_;

  f32x8 acc[4] = {};
  for (int k0 = 0; k0 < NQ_; k0 += 32) {
    const bf16x16 a = ld_frag(orow + k0 + (lhalf << 3),
                              orow + k0 + 16 + (lhalf << 3));
#pragma unroll
    for (int t = 0; t < 4; ++t) {
      const __bf16* wrow =
          Wob + (size_t)(c0 + t * 16 + col) * NQ_ + k0 + (lhalf << 3);
      acc[t] = wmma_bf16(a, ld_frag(wrow, wrow + 16), acc[t]);
    }
  }

  float* outb = out + (size_t)b * C_ * L_;
#pragma unroll
  for (int t = 0; t < 4; ++t) {
#pragma unroll
    for (int v = 0; v < 8; ++v) {
      const int l = m0 + v + (lhalf << 3);
      outb[(size_t)(c0 + t * 16 + col) * L_ + l] = acc[t][v];
    }
  }
}

// ---------------------------------------------------------------------------
extern "C" void kernel_launch(void* const* d_in, const int* in_sizes, int n_in,
                              void* d_out, int out_size, void* d_ws,
                              size_t ws_size, hipStream_t stream) {
  (void)in_sizes; (void)n_in; (void)out_size; (void)ws_size;

  const float* x   = (const float*)d_in[0];
  const float* Wq  = (const float*)d_in[1];
  const float* Wkv = (const float*)d_in[2];
  const float* Wo  = (const float*)d_in[3];
  float* out = (float*)d_out;

  char* ws = (char*)d_ws;
  size_t off = 0;
  __bf16* Xt   = (__bf16*)(ws + off); off += (size_t)B_ * L_ * C_ * 2;     // 8 MB
  __bf16* Wqb  = (__bf16*)(ws + off); off += (size_t)NQ_ * C_ * 2;         // 512 KB
  __bf16* Wkvb = (__bf16*)(ws + off); off += (size_t)NKV_ * C_ * 2;        // 256 KB
  __bf16* Wob  = (__bf16*)(ws + off); off += (size_t)C_ * NQ_ * 2;         // 512 KB
  __bf16* Qb   = (__bf16*)(ws + off); off += (size_t)B_ * H_ * L_ * DH_ * 2;   // 8 MB
  __bf16* Kb   = (__bf16*)(ws + off); off += (size_t)B_ * KVH_ * L_ * DH_ * 2; // 2 MB
  __bf16* Vt   = (__bf16*)(ws + off); off += (size_t)B_ * KVH_ * L_ * DH_ * 2; // 2 MB
  __bf16* Ob   = (__bf16*)(ws + off); off += (size_t)B_ * L_ * NQ_ * 2;        // 8 MB

  // --- prep: transpose x to bf16 [b][l][c]; convert weights to bf16 ---
  transpose_x_kernel<<<dim3(L_ / 32, C_ / 32, B_), dim3(32, 8), 0, stream>>>(x, Xt);
  cvt_bf16_kernel<<<(NQ_ * C_ / 4 + 255) / 256, 256, 0, stream>>>(Wq, Wqb, NQ_ * C_ / 4);
  cvt_bf16_kernel<<<(NKV_ * C_ / 4 + 255) / 256, 256, 0, stream>>>(Wkv, Wkvb, NKV_ * C_ / 4);
  cvt_bf16_kernel<<<(C_ * NQ_ / 4 + 255) / 256, 256, 0, stream>>>(Wo, Wob, C_ * NQ_ / 4);

  // --- QKV + RoPE: grid 256 x 3 x 2, 4 waves/block, 16x64 tile per wave ---
  qkv_rope_kernel<<<dim3(L_ / 16, (NQ_ + NKV_) / 256, B_), 128, 0, stream>>>(
      Xt, Wqb, Wkvb, Qb, Kb, Vt);

  // --- flash attention: grid 64 x 8 x 2 ---
  flash_attn_kernel<<<dim3(L_ / 64, H_, B_), 128, 0, stream>>>(Qb, Kb, Vt, Ob);

  // --- output projection: grid 256 x 2 x 2 ---
  out_proj_kernel<<<dim3(L_ / 16, C_ / 256, B_), 128, 0, stream>>>(Ob, Wob, out);
}
